// Learnable_Block_22084721836772
// MI455X (gfx1250) — matrix operand order repaired and checked by reading, not compiled, not run on hardware
//
#include <hip/hip_runtime.h>

typedef __attribute__((ext_vector_type(16))) __bf16 v16bf;
typedef __attribute__((ext_vector_type(8)))  float  v8f;
typedef int vsi4 __attribute__((vector_size(16)));   // matches builtin's int4 param

#define H 320
#define W 320
#define NB 2
#define F 48
#define PLANES 96            // [ar(48) | ai(48)]
#define TILE_W 64
#define HALO_W 66
#define PER_LAYER_PACK (27*6*32*16)   // 82944 bf16 per layer

#if defined(__AMDGCN__) && __has_builtin(__builtin_amdgcn_global_load_async_to_lds_b128) && __has_builtin(__builtin_amdgcn_s_wait_asynccnt)
#define USE_ASYNC_LDS 1
#else
#define USE_ASYNC_LDS 0
#endif

__device__ __forceinline__ float sigma_f(float x) {
    float ax = __builtin_fabsf(x);
    return (ax > 0.001f) ? __builtin_fmaxf(x, 0.0f)
                         : (x * x * 250.0f + 0.5f * x + 0.00025f);
}

// ---------------------------------------------------------------------------
// Pack complex-expanded weights [[wr,-wi],[wi,wr]] (96 x 864) into WMMA
// A-fragment order for v_wmma_f32_16x16x32_bf16.
// Chunk kc = tap*3 + pc; covers K = tap*96 + pc*32 ..+31.
// A layout (contiguous k, matches B): lanes 0-15 -> Klocal 0..15; lanes 16-31 -> 16..31.
// ---------------------------------------------------------------------------
__global__ void pack_weights_kernel(const float* __restrict__ w1r, const float* __restrict__ w1i,
                                    const float* __restrict__ w2r, const float* __restrict__ w2i,
                                    __bf16* __restrict__ pW1, __bf16* __restrict__ pW2) {
    int idx = blockIdx.x * 256 + threadIdx.x;
    if (idx >= 2 * PER_LAYER_PACK) return;
    int layer = idx / PER_LAYER_PACK;
    int r = idx % PER_LAYER_PACK;           // r = ((kc*6 + mt)*32 + L)*16 + e
    int e  = r & 15;
    int t  = r >> 4;
    int L  = t & 31;
    t >>= 5;
    int mt = t % 6;
    int kc = t / 6;

    int m      = mt * 16 + (L & 15);                       // output row 0..95
    int klocal = e + ((L >> 4) * 16);                      // contiguous k per lane half
    int tap    = kc / 3;
    int pc     = kc % 3;
    int plane  = pc * 32 + klocal;                         // 0..95

    const float* wr = layer ? w2r : w1r;
    const float* wi = layer ? w2i : w1i;
    int f   = (m < F) ? m : (m - F);
    int cin = (plane < F) ? plane : (plane - F);
    int wofs = (f * F + cin) * 9 + tap;                    // [F][C][3][3] row-major

    float val;
    if (m < F) val = (plane < F) ?  wr[wofs] : -wi[wofs];  // real row: [wr | -wi]
    else       val = (plane < F) ?  wi[wofs] :  wr[wofs];  // imag row: [wi |  wr]

    (layer ? pW2 : pW1)[r] = (__bf16)val;
}

// ---------------------------------------------------------------------------
// Layer 0: Cin=1 complex conv (tiny FLOPs) -> fp32 conv output + bf16 activations
// act buffer layout: [b][y][x][plane] plane-fastest (96 bf16 per pixel).
// ---------------------------------------------------------------------------
__global__ void layer0_kernel(const float* __restrict__ xr, const float* __restrict__ xi,
                              const float* __restrict__ w0r, const float* __restrict__ w0i,
                              const float* __restrict__ b0r, const float* __restrict__ b0i,
                              float* __restrict__ out0, __bf16* __restrict__ actOut) {
    int idx = blockIdx.x * 256 + threadIdx.x;
    if (idx >= NB * F * H * W) return;
    int x = idx % W; int t = idx / W;
    int y = t % H;   t /= H;
    int f = t % F;   int b = t / F;

    float r = b0r[f], i = b0i[f];
    const float* xrb = xr + (size_t)b * H * W;
    const float* xib = xi + (size_t)b * H * W;
#pragma unroll
    for (int dy = 0; dy < 3; ++dy) {
        int gy = y + dy - 1;
        if (gy < 0 || gy >= H) continue;
#pragma unroll
        for (int dx = 0; dx < 3; ++dx) {
            int gx = x + dx - 1;
            if (gx < 0 || gx >= W) continue;
            float ar = xrb[gy * W + gx], ai = xib[gy * W + gx];
            float wr = w0r[f * 9 + dy * 3 + dx], wi = w0i[f * 9 + dy * 3 + dx];
            r += ar * wr - ai * wi;
            i += ar * wi + ai * wr;
        }
    }
    size_t o = ((((size_t)b * F + f) * H + y) * W + x) * 2;
    __builtin_nontemporal_store(r, &out0[o + 0]);
    __builtin_nontemporal_store(i, &out0[o + 1]);
    size_t ao = (((size_t)b * H + y) * W + x) * PLANES;
    actOut[ao + f]     = (__bf16)sigma_f(r);
    actOut[ao + F + f] = (__bf16)sigma_f(i);
}

// ---------------------------------------------------------------------------
// Layers 1/2: implicit-GEMM complex conv via WMMA bf16.
// Block: 128 threads (4 waves), covers 2 rows x 64 pixels, all 96 out rows.
// Wave (row, col-half) owns 2 N-tiles of 16 pixels and all 6 M-tiles:
// 12 WMMAs share each set of 6 A-fragments -> halved weight-load traffic.
// K = 864 in 27 chunks of 32 planes at one tap.
// ---------------------------------------------------------------------------
__global__ void __launch_bounds__(128)
cconv_gemm_kernel(const __bf16* __restrict__ actIn,   // [B][H][W][96] bf16
                  const __bf16* __restrict__ pW,      // packed A-fragments
                  const float*  __restrict__ br, const float* __restrict__ bi,
                  float* __restrict__ outConv,        // slot base in d_out
                  __bf16* __restrict__ actOut)        // next-layer acts or nullptr
{
    const int b     = blockIdx.z;
    const int y0    = blockIdx.y * 2;
    const int xbase = blockIdx.x * TILE_W;
    const int tid   = threadIdx.x;

    __shared__ __attribute__((aligned(32))) __bf16 s_act[4 * HALO_W * PLANES]; // 50.7 KB
    __shared__ float s_bias[PLANES];

    if (tid < PLANES) s_bias[tid] = (tid < F) ? br[tid] : bi[tid - F];

    // Stage halo: 4 rows x 66 cols, each site = 96 contiguous bf16 (192 B).
    for (int s = tid; s < 4 * HALO_W; s += 128) {
        int row = s / HALO_W, col = s % HALO_W;
        int gy = y0 + row - 1;
        int gx = xbase + col - 1;
        __bf16* dst = &s_act[s * PLANES];
        if (gy >= 0 && gy < H && gx >= 0 && gx < W) {
            const __bf16* src = actIn + (((size_t)b * H + gy) * W + gx) * PLANES;
#if USE_ASYNC_LDS
#pragma unroll
            for (int q = 0; q < 12; ++q) {
                __builtin_amdgcn_global_load_async_to_lds_b128(
                    (vsi4*)(src + q * 8),
                    (__attribute__((address_space(3))) vsi4*)(dst + q * 8),
                    0, 0);
            }
#else
            const uint4* s4 = (const uint4*)src;
            uint4* d4 = (uint4*)dst;
#pragma unroll
            for (int q = 0; q < 12; ++q) d4[q] = s4[q];
#endif
        } else {
            uint4 z = make_uint4(0u, 0u, 0u, 0u);
            uint4* d4 = (uint4*)dst;
#pragma unroll
            for (int q = 0; q < 12; ++q) d4[q] = z;
        }
    }
#if USE_ASYNC_LDS
    __builtin_amdgcn_s_wait_asynccnt(0);
#endif
    __syncthreads();

    const int wv   = tid >> 5;
    const int lane = tid & 31;
    const int ln16 = lane & 15;
    const int hi   = lane >> 4;           // selects K 0-15 vs 16-31 of the chunk
    const int rrow = wv >> 1;             // 0/1: which output row
    const int ch   = wv & 1;              // 0/1: which 32-pixel column half
    const int col0 = ch * 32 + ln16;      // pixel (tile 0) within 64-wide tile
    const int col1 = col0 + 16;           // pixel (tile 1)

    v8f acc[12];
#pragma unroll
    for (int t = 0; t < 12; ++t) acc[t] = (v8f){0,0,0,0,0,0,0,0};

#pragma unroll 3
    for (int kc = 0; kc < 27; ++kc) {
        const int tap = kc / 3;
        const int pc  = kc % 3;
        const int dy  = tap / 3, dx = tap % 3;
        const int pofs = pc * 32 + hi * 16;
        const int rofs = (rrow + dy) * HALO_W;
        // B fragments: lane's pixel column, 16 consecutive planes (32 B in LDS)
        v16bf bf0 = *(const v16bf*)&s_act[(rofs + col0 + dx) * PLANES + pofs];
        v16bf bf1 = *(const v16bf*)&s_act[(rofs + col1 + dx) * PLANES + pofs];
        const __bf16* ap = pW + ((size_t)(kc * 6) * 32 + lane) * 16;
#pragma unroll
        for (int mt = 0; mt < 6; ++mt) {
            v16bf afrag = *(const v16bf*)(ap + (size_t)mt * 32 * 16);
            acc[mt * 2 + 0] = __builtin_amdgcn_wmma_f32_16x16x32_bf16(
                false, afrag, false, bf0, (short)0, acc[mt * 2 + 0], false, false);
            acc[mt * 2 + 1] = __builtin_amdgcn_wmma_f32_16x16x32_bf16(
                false, afrag, false, bf1, (short)0, acc[mt * 2 + 1], false, false);
        }
    }

    // Writeback: C/D layout -> VGPR r, lanes 0-15: M=r, N=lane; lanes 16-31: M=r+8.
    const int y = y0 + rrow;
#pragma unroll
    for (int mt = 0; mt < 6; ++mt) {
#pragma unroll
        for (int j = 0; j < 2; ++j) {
            const int px = xbase + ch * 32 + j * 16 + ln16;
#pragma unroll
            for (int r = 0; r < 8; ++r) {
                int m = mt * 16 + r + hi * 8;
                float v = acc[mt * 2 + j][r] + s_bias[m];
                int f = (m < F) ? m : (m - F);
                int c = (m < F) ? 0 : 1;
                size_t o = ((((size_t)b * F + f) * H + y) * W + px) * 2 + c;
                __builtin_nontemporal_store(v, &outConv[o]);
                if (actOut) {
                    actOut[(((size_t)b * H + y) * W + px) * PLANES + m] = (__bf16)sigma_f(v);
                }
            }
        }
    }
}

// ---------------------------------------------------------------------------
extern "C" void kernel_launch(void* const* d_in, const int* in_sizes, int n_in,
                              void* d_out, int out_size, void* d_ws, size_t ws_size,
                              hipStream_t stream) {
    const float* x_r = (const float*)d_in[0];
    const float* x_i = (const float*)d_in[1];
    const float* w0r = (const float*)d_in[2];
    const float* w0i = (const float*)d_in[3];
    const float* b0r = (const float*)d_in[4];
    const float* b0i = (const float*)d_in[5];
    const float* w1r = (const float*)d_in[6];
    const float* w1i = (const float*)d_in[7];
    const float* b1r = (const float*)d_in[8];
    const float* b1i = (const float*)d_in[9];
    const float* w2r = (const float*)d_in[10];
    const float* w2i = (const float*)d_in[11];
    const float* b2r = (const float*)d_in[12];
    const float* b2i = (const float*)d_in[13];
    float* out = (float*)d_out;

    // Workspace layout
    const size_t packBytes = (size_t)PER_LAYER_PACK * 2;         // 165,888 B each
    const size_t actBytes  = (size_t)NB * H * W * PLANES * 2;    // 39,321,600 B each
    char* ws = (char*)d_ws;
    __bf16* pW1 = (__bf16*)(ws);
    __bf16* pW2 = (__bf16*)(ws + packBytes);
    __bf16* A1  = (__bf16*)(ws + 2 * packBytes);
    __bf16* A2  = (__bf16*)(ws + 2 * packBytes + actBytes);

    const size_t slot = (size_t)NB * F * H * W * 2;              // 19,660,800 floats

    pack_weights_kernel<<<(2 * PER_LAYER_PACK + 255) / 256, 256, 0, stream>>>(
        w1r, w1i, w2r, w2i, pW1, pW2);

    layer0_kernel<<<(NB * F * H * W + 255) / 256, 256, 0, stream>>>(
        x_r, x_i, w0r, w0i, b0r, b0i, out, A1);

    dim3 g(W / TILE_W, H / 2, NB);
    cconv_gemm_kernel<<<g, 128, 0, stream>>>(A1, pW1, b1r, b1i, out + slot, A2);
    cconv_gemm_kernel<<<g, 128, 0, stream>>>(A2, pW2, b2r, b2i, out + 2 * slot, (__bf16*)nullptr);
}
